// Attention_24919400252009
// MI455X (gfx1250) — matrix operand order repaired and checked by reading, not compiled, not run on hardware
//
#include <hip/hip_runtime.h>
#include <hip/hip_bf16.h>
#include <math.h>

// ---------------------------------------------------------------------------
// Fused MHA for MI455X (gfx1250, wave32, WMMA).
// Roofline: ~13 GFLOP vs 256 MiB attn_probs fp32 stream-out -> memory bound.
// => f16 WMMA (f32 accum) everywhere, 128-bit NT stores for the probs stream,
//    V stored transposed so PV B-fragments are two 16B loads along k.
// ---------------------------------------------------------------------------

typedef __attribute__((ext_vector_type(16))) _Float16 v16h;
typedef __attribute__((ext_vector_type(8)))  _Float16 v8h;
typedef __attribute__((ext_vector_type(4)))  _Float16 v4h;
typedef __attribute__((ext_vector_type(8)))  float    v8f;
typedef __attribute__((ext_vector_type(4)))  float    v4f;
typedef __attribute__((ext_vector_type(4)))  int      v4i;

constexpr int Bc = 2, Gc = 4, Qc = 1024, Kc = 1024;
constexpr int Cc = 256, Hc = 8, HDc = 32;

__device__ __forceinline__ v8f wmma_f16(v16h a, v16h b, v8f c) {
  // v_wmma_f32_16x16x32_f16  (neg_a, A, neg_b, B, c_mod, C, reuse_a, reuse_b)
  return __builtin_amdgcn_wmma_f32_16x16x32_f16(false, a, false, b, (short)0, c,
                                                false, false);
}
__device__ __forceinline__ v16h cat8(v8h lo, v8h hi) {
  return __builtin_shufflevector(lo, hi, 0,1,2,3,4,5,6,7,8,9,10,11,12,13,14,15);
}

// ---- weight fp32 -> f16 ----------------------------------------------------
__global__ void cvt_w_f16(const float* __restrict__ src,
                          _Float16* __restrict__ dst, int n) {
  int i = blockIdx.x * 256 + threadIdx.x;
  if (i < n) dst[i] = (_Float16)src[i];
}

// ---- Y[M,256] = X[M,256] @ W16^T + bias  (M = 8192 rows) -------------------
// One 16x16 output tile per wave, K-loop = 8 WMMA(16x16x32) steps.
// X_F16: X already f16 (attention output); else fp32 converted on load.
// OUT_F32: write fp32 (final output); STORE_T: write f16 transposed per
// (b,g) slab: Y[(bg*256+col)*1024 + key]  (used for V).
template<bool X_F16, bool OUT_F32, bool STORE_T>
__global__ void __launch_bounds__(256) gemm256(
    const void* __restrict__ Xv, const _Float16* __restrict__ W,
    const float* __restrict__ bias, void* __restrict__ Yv) {
  const int lane = threadIdx.x & 31, wave = threadIdx.x >> 5;
  const int half = lane >> 4, ln = lane & 15;
  const int tile = blockIdx.x * 8 + wave;          // 8192 tiles total
  const int mt = tile >> 4, nt = tile & 15;
  const int row = mt * 16 + ln;
  const int col = nt * 16 + ln;
  const _Float16* wrow = W + (size_t)col * Cc;     // B[k][n] = W[n][k]

  v8f acc = {};
  if (X_F16) {
    const _Float16* xrow = (const _Float16*)Xv + (size_t)row * Cc;
    #pragma unroll
    for (int k0 = 0; k0 < Cc; k0 += 32) {
      int ka = k0 + half * 8;
      v16h a   = cat8(*(const v8h*)(xrow + ka), *(const v8h*)(xrow + ka + 16));
      v16h bfr = cat8(*(const v8h*)(wrow + ka), *(const v8h*)(wrow + ka + 16));
      acc = wmma_f16(a, bfr, acc);
    }
  } else {
    const float* xrow = (const float*)Xv + (size_t)row * Cc;
    #pragma unroll
    for (int k0 = 0; k0 < Cc; k0 += 32) {
      int ka = k0 + half * 8;
      v8f x0 = *(const v8f*)(xrow + ka);
      v8f x1 = *(const v8f*)(xrow + ka + 16);
      v16h a = cat8(__builtin_convertvector(x0, v8h),
                    __builtin_convertvector(x1, v8h));
      v16h bfr = cat8(*(const v8h*)(wrow + ka), *(const v8h*)(wrow + ka + 16));
      acc = wmma_f16(a, bfr, acc);
    }
  }

  const float bcol = bias[col];
  if (OUT_F32) {
    float* Y = (float*)Yv;
    #pragma unroll
    for (int r = 0; r < 8; r++) {
      int m = mt * 16 + half * 8 + r;
      Y[(size_t)m * Cc + col] = acc[r] + bcol;
    }
  } else if (STORE_T) {
    _Float16* Y = (_Float16*)Yv;
    int mbase = mt * 16 + half * 8;
    int bg = mbase >> 10, key = mbase & 1023;      // 8 consecutive keys
    v8h o;
    #pragma unroll
    for (int r = 0; r < 8; r++) o[r] = (_Float16)(acc[r] + bcol);
    *(v8h*)(Y + ((size_t)bg * Cc + col) * Kc + key) = o;
  } else {
    _Float16* Y = (_Float16*)Yv;
    #pragma unroll
    for (int r = 0; r < 8; r++) {
      int m = mt * 16 + half * 8 + r;
      Y[(size_t)m * Cc + col] = (_Float16)(acc[r] + bcol);
    }
  }
}

// ---- fused attention: scores -> masked softmax -> probs out -> P@V ---------
// One block (8 waves) per (b,g,h, 16-row query block). 4096 blocks.
__global__ void __launch_bounds__(256) attn_kernel(
    const _Float16* __restrict__ q16, const _Float16* __restrict__ k16,
    const _Float16* __restrict__ vT16, const int* __restrict__ mask,
    float* __restrict__ probs_out, _Float16* __restrict__ ao16) {
  __shared__ float    S[16 * Kc];        // 64 KB raw/exp scores (fp32)
  __shared__ _Float16 P[16 * Kc];        // 32 KB unnormalized exp (f16)
  __shared__ float    red0[256];
  __shared__ float    red1[256];
  __shared__ float    Oacc[8][2][256];   // 16 KB per-wave PV partials
  __shared__ float    rowInv[16];

  const int tid  = threadIdx.x;
  const int lane = tid & 31, wave = tid >> 5;
  const int half = lane >> 4, ln = lane & 15;

  const int bid = blockIdx.x;
  const int rb = bid & 63;
  const int h  = (bid >> 6) & 7;
  const int g  = (bid >> 9) & 3;
  const int b  = bid >> 11;
  const int bg = b * Gc + g;

  const _Float16* qh = q16 + ((size_t)bg * Qc) * Cc + h * HDc;
  const _Float16* kh = k16 + ((size_t)bg * Kc) * Cc + h * HDc;

  // ---- Phase A: S = (q_h @ k_h^T) * scale  (HD=32 -> single WMMA k-pass)
  const float scale = 0.17677669529663687f;  // 1/sqrt(32)
  v16h afrag;
  {
    const _Float16* qp = qh + (size_t)(rb * 16 + ln) * Cc + half * 8;
    afrag = cat8(*(const v8h*)qp, *(const v8h*)(qp + 16));
  }
  #pragma unroll
  for (int i = 0; i < 8; i++) {
    int nt = wave * 8 + i;                                  // key tile 0..63
    const _Float16* kp = kh + (size_t)(nt * 16 + ln) * Cc + half * 8;
    v16h bfr = cat8(*(const v8h*)kp, *(const v8h*)(kp + 16));
    v8f z = {};
    v8f s = wmma_f16(afrag, bfr, z);
    int colb = nt * 16 + ln;
    #pragma unroll
    for (int r = 0; r < 8; r++)
      S[(r + half * 8) * Kc + colb] = s[r] * scale;
  }
  __syncthreads();

  // ---- Phase B: mask + softmax (fp32), all traffic 128-bit wide.
  {
    const int row = tid >> 4, seg = tid & 15;               // 16 thr per row
    const int qrow = rb * 16 + row;
    const v4i* mrow4 =
        (const v4i*)(mask + ((size_t)b * Qc + qrow) * Kc + seg * 64);
    v4f* srow4 = (v4f*)(S + row * Kc + seg * 64);

    float mx = -1e30f;
    #pragma unroll
    for (int j = 0; j < 16; j++) {
      v4i m = mrow4[j];
      v4f v = srow4[j];
      #pragma unroll
      for (int c = 0; c < 4; c++) {
        v[c] = m[c] ? v[c] : -1e30f;
        mx = fmaxf(mx, v[c]);
      }
      srow4[j] = v;
    }
    red0[tid] = mx;
    __syncthreads();
    float rowmax = -1e30f;
    #pragma unroll
    for (int t = 0; t < 16; t++) rowmax = fmaxf(rowmax, red0[row * 16 + t]);

    float sum = 0.f;
    v4h* prow4 = (v4h*)(P + row * Kc + seg * 64);
    #pragma unroll
    for (int j = 0; j < 16; j++) {
      v4f v = srow4[j];
      v4f e;
      #pragma unroll
      for (int c = 0; c < 4; c++) e[c] = __expf(v[c] - rowmax);
      srow4[j] = e;
      prow4[j] = __builtin_convertvector(e, v4h);  // normalization folded into PV
      sum += e[0] + e[1] + e[2] + e[3];
    }
    red1[tid] = sum;
    __syncthreads();
    float rowsum = 0.f;
    #pragma unroll
    for (int t = 0; t < 16; t++) rowsum += red1[row * 16 + t];
    float inv = 1.0f / rowsum;
    if (seg == 0) rowInv[row] = inv;

    // 256 MiB probs stream: 128-bit non-temporal stores, fully coalesced.
    v4f* gp4 = (v4f*)(probs_out +
                      (((size_t)(bg * Hc + h) * Qc + qrow) * Kc) + seg * 64);
    #pragma unroll
    for (int j = 0; j < 16; j++) {
      v4f p = srow4[j];
      #pragma unroll
      for (int c = 0; c < 4; c++) p[c] *= inv;
      __builtin_nontemporal_store(p, gp4 + j);
    }
  }
  __syncthreads();

  // ---- Phase C: out = (P @ v_h) * rowInv ; wave w reads keys [w*128, w*128+256)
  {
    const _Float16* vt = vT16 + ((size_t)bg * Cc + h * HDc) * Kc;
    v8f acc0 = {}, acc1 = {};
    #pragma unroll
    for (int ks = 0; ks < 8; ks++) {
      int kk = wave * 128 + ks * 32;
      const _Float16* pp = P + ln * Kc + kk + half * 8;
      v16h a = cat8(*(const v8h*)pp, *(const v8h*)(pp + 16));
      int kb = kk + half * 8;
      v16h b0 = cat8(*(const v8h*)(vt + (size_t)ln * Kc + kb), *(const v8h*)(vt + (size_t)ln * Kc + kb + 16));        // d = 0..15
      v16h b1 = cat8(*(const v8h*)(vt + (size_t)(16 + ln) * Kc + kb), *(const v8h*)(vt + (size_t)(16 + ln) * Kc + kb + 16)); // d = 16..31
      acc0 = wmma_f16(a, b0, acc0);
      acc1 = wmma_f16(a, b1, acc1);
    }
    #pragma unroll
    for (int r = 0; r < 8; r++) {
      Oacc[wave][0][(r + half * 8) * 16 + ln] = acc0[r];
      Oacc[wave][1][(r + half * 8) * 16 + ln] = acc1[r];
    }
  }
  __syncthreads();

  for (int e = tid; e < 512; e += 256) {
    int nt = e >> 8, pos = e & 255, m = pos >> 4, n = pos & 15;
    float s = 0.f;
    #pragma unroll
    for (int w = 0; w < 8; w++) s += Oacc[w][nt][pos];
    s *= rowInv[m];
    ao16[((size_t)bg * Qc + rb * 16 + m) * Cc + h * HDc + nt * 16 + n] =
        (_Float16)s;
  }
}

// ---------------------------------------------------------------------------
extern "C" void kernel_launch(void* const* d_in, const int* in_sizes, int n_in,
                              void* d_out, int out_size, void* d_ws,
                              size_t ws_size, hipStream_t stream) {
  (void)in_sizes; (void)n_in; (void)out_size; (void)ws_size;
  const float* inputs_q  = (const float*)d_in[0];
  const float* inputs_kv = (const float*)d_in[1];
  const int*   mask      = (const int*)d_in[2];
  const float* Wq = (const float*)d_in[3];
  const float* bq = (const float*)d_in[4];
  const float* Wk = (const float*)d_in[5];
  const float* bk = (const float*)d_in[6];
  const float* Wv = (const float*)d_in[7];
  const float* bv = (const float*)d_in[8];
  const float* Wo = (const float*)d_in[9];
  const float* bo = (const float*)d_in[10];

  // workspace layout (~16.5 MiB total)
  char* ws = (char*)d_ws;
  _Float16* wq16 = (_Float16*)(ws);
  _Float16* wk16 = (_Float16*)(ws + (128 << 10));
  _Float16* wv16 = (_Float16*)(ws + (256 << 10));
  _Float16* wo16 = (_Float16*)(ws + (384 << 10));
  _Float16* q16  = (_Float16*)(ws + (512 << 10));
  _Float16* k16  = (_Float16*)(ws + (512 << 10) + (4 << 20));
  _Float16* vT16 = (_Float16*)(ws + (512 << 10) + (8 << 20));
  _Float16* ao16 = (_Float16*)(ws + (512 << 10) + (12 << 20));

  float* out_final = (float*)d_out;                         // [B,G,Q,256]
  float* out_probs = out_final + (size_t)Bc * Gc * Qc * Cc; // [B,G,H,Q,K]

  cvt_w_f16<<<256, 256, 0, stream>>>(Wq, wq16, Cc * Cc);
  cvt_w_f16<<<256, 256, 0, stream>>>(Wk, wk16, Cc * Cc);
  cvt_w_f16<<<256, 256, 0, stream>>>(Wv, wv16, Cc * Cc);
  cvt_w_f16<<<256, 256, 0, stream>>>(Wo, wo16, Cc * Cc);

  gemm256<false, false, false><<<1024, 256, 0, stream>>>(inputs_q,  wq16, bq, q16);
  gemm256<false, false, false><<<1024, 256, 0, stream>>>(inputs_kv, wk16, bk, k16);
  gemm256<false, false, true ><<<1024, 256, 0, stream>>>(inputs_kv, wv16, bv, vT16);

  attn_kernel<<<Bc * Gc * Hc * (Qc / 16), 256, 0, stream>>>(
      q16, k16, vT16, mask, out_probs, ao16);

  gemm256<true, true, false><<<1024, 256, 0, stream>>>(ao16, wo16, bo, out_final);
}
